// MultiNadwatSpatialConv_66322884985177
// MI455X (gfx1250) — matrix operand order repaired and checked by reading, not compiled, not run on hardware
//
#include <hip/hip_runtime.h>

typedef __attribute__((ext_vector_type(16))) _Float16 v16h;
typedef __attribute__((ext_vector_type(8)))  float    v8f;
typedef __attribute__((ext_vector_type(4)))  float    v4f;
typedef __attribute__((ext_vector_type(2)))  float    v2f;

#define B_   2
#define N_   4096
#define M_   4096
#define C_   64
#define MT_  (M_ / 32)   // 128 m-tiles of 32
#define NT_  (N_ / 16)   // 256 n-tiles of 16

// workspace layout (bytes)
#define FEA_BYTES  ((size_t)B_ * M_ * C_ * 2)        // f16 packed B-layout: 1 MB
#define YA_OFF     (FEA_BYTES)                       // float4 per m: (-2y0,-2y1,-2y2,0)
#define YA_BYTES   ((size_t)B_ * M_ * 4 * 4)         // 128 KB
#define YAUX_OFF   (YA_OFF + YA_BYTES)               // float2 per m: (|y|^2, w)

// -25/ln(2): e = exp(-25*d2) = exp2(KC1*d2); e^4, e^16 give the other scales
#define KC1 (-36.06737602222409f)

// ---------------- pass 1a: pack y coords / |y|^2 / weight ----------------
__global__ void pack_y_kernel(const float* __restrict__ y,
                              const float* __restrict__ yw,
                              float* __restrict__ yA,
                              float* __restrict__ yaux) {
    int i = blockIdx.x * blockDim.x + threadIdx.x;   // 0 .. B*M-1
    if (i >= B_ * M_) return;
    float y0 = y[i * 3 + 0], y1 = y[i * 3 + 1], y2 = y[i * 3 + 2];
    yA[i * 4 + 0] = -2.f * y0;
    yA[i * 4 + 1] = -2.f * y1;
    yA[i * 4 + 2] = -2.f * y2;
    yA[i * 4 + 3] = 0.f;
    yaux[i * 2 + 0] = y0 * y0 + y1 * y1 + y2 * y2;
    yaux[i * 2 + 1] = yw[i];
}

// ---------------- pass 1b: y_fea f32 -> f16 in WMMA B-matrix layout ----------------
// B matrix (f16, 32x16): lanes 0-15 = cols, K=0..7,16..23 packed 2-per-VGPR;
// lanes 16-31 = cols, K=8..15,24..31.  Flat: (((b*MT+mt)*4+ct)*32+lane)*16 + j, k = (j&7) + 8*(lane>>4) + 16*(j>>3)
__global__ void pack_fea_kernel(const float* __restrict__ fea,
                                _Float16* __restrict__ feaB) {
    int i = blockIdx.x * blockDim.x + threadIdx.x;   // 0 .. B*M*C-1
    if (i >= B_ * M_ * C_) return;
    int j    = i & 15;
    int lane = (i >> 4) & 31;
    int ct   = (i >> 9) & 3;
    int mt   = (i >> 11) & (MT_ - 1);
    int b    = i >> 18;
    int col  = lane & 15;
    int h    = lane >> 4;
    int k    = (j & 7) + 8 * h + 16 * (j >> 3);
    feaB[i] = (_Float16)fea[((size_t)(b * M_ + mt * 32 + k)) * C_ + ct * 16 + col];
}

// ---------------- pass 2: fused distance/exp/NW kernel ----------------
__global__ __launch_bounds__(128) void nadwat_main_kernel(
    const float* __restrict__ x, const _Float16* __restrict__ feaB,
    const float* __restrict__ yA, const float* __restrict__ yaux,
    float* __restrict__ out) {
    const int wave = threadIdx.x >> 5;
    const int lane = threadIdx.x & 31;
    const int tile = blockIdx.x * 4 + wave;          // 0 .. B*NT-1
    const int b    = tile / NT_;
    const int n0   = (tile % NT_) * 16;
    const int col  = lane & 15;
    const int h    = lane >> 4;

    // x row for this lane's n column (B operand of the f32 WMMA: B[k][n] = x[n][k])
    const float* xp = x + (size_t)(b * N_ + n0 + col) * 3;
    const float xx = xp[0], xy = xp[1], xz = xp[2];
    const float x2 = xx * xx + xy * xy + xz * xz;
    v2f bx;
    bx.x = h ? xz : xx;          // dims {0,1} for lanes 0-15, {2,3(=0)} for 16-31
    bx.y = h ? 0.f : xy;

    v8f acc0 = {}, acc1 = {}, acc2 = {}, acc3 = {};
    float den = 0.f;

    const float* yAb   = yA   + (size_t)b * M_ * 4;
    const float* yauxb = yaux + (size_t)b * M_ * 2;
    const v16h*  fbb   = (const v16h*)feaB + (size_t)b * MT_ * 4 * 32 + lane;

    for (int mt = 0; mt < MT_; ++mt) {
        const int m0 = mt * 32;
        // A operand of f32 WMMA: A[m][k] = -2*y[m][k]; lane row = col, dims 2h,2h+1
        v2f a0 = *(const v2f*)(yAb + (size_t)(m0 + col) * 4 + 2 * h);
        v2f a1 = *(const v2f*)(yAb + (size_t)(m0 + 16 + col) * 4 + 2 * h);
        // |y|^2 and weights for this lane's 8 m-values per sub-tile (contiguous 64B)
        const v4f* qp0 = (const v4f*)(yauxb + (size_t)(m0 + 8 * h) * 2);
        const v4f* qp1 = (const v4f*)(yauxb + (size_t)(m0 + 16 + 8 * h) * 2);
        v4f q0 = qp0[0], q1 = qp0[1], q2 = qp0[2], q3 = qp0[3];
        v4f r0 = qp1[0], r1 = qp1[1], r2 = qp1[2], r3 = qp1[3];
        float y2a[8] = {q0.x, q0.z, q1.x, q1.z, q2.x, q2.z, q3.x, q3.z};
        float ywa[8] = {q0.y, q0.w, q1.y, q1.w, q2.y, q2.w, q3.y, q3.w};
        float y2b[8] = {r0.x, r0.z, r1.x, r1.z, r2.x, r2.z, r3.x, r3.z};
        float ywb[8] = {r0.y, r0.w, r1.y, r1.w, r2.y, r2.w, r3.y, r3.w};

        v8f c0, c1;
        #pragma unroll
        for (int r = 0; r < 8; ++r) { c0[r] = x2 + y2a[r]; c1[r] = x2 + y2b[r]; }

        // D = (-2y)·x + (x^2 + y^2) = d^2 ; lane=n, VGPR r = m (matches f16 A layout!)
        v8f d0 = __builtin_amdgcn_wmma_f32_16x16x4_f32(false, a0, false, bx,
                                                       (short)0, c0, false, false);
        v8f d1 = __builtin_amdgcn_wmma_f32_16x16x4_f32(false, a1, false, bx,
                                                       (short)0, c1, false, false);

        v16h aK;
        #pragma unroll
        for (int r = 0; r < 8; ++r) {
            float t  = fmaxf(d0[r], 0.f);
            float e  = __builtin_amdgcn_exp2f(KC1 * t);     // exp(-25 d2)
            float e2 = e * e, e4 = e2 * e2, e8 = e4 * e4, e16 = e8 * e8;
            float K  = fmaf(0.3f, e16, fmaf(0.3f, e4, 0.4f * e));
            float kw = K * ywa[r];
            den += kw;
            aK[r] = (_Float16)kw;
        }
        #pragma unroll
        for (int r = 0; r < 8; ++r) {
            float t  = fmaxf(d1[r], 0.f);
            float e  = __builtin_amdgcn_exp2f(KC1 * t);
            float e2 = e * e, e4 = e2 * e2, e8 = e4 * e4, e16 = e8 * e8;
            float K  = fmaf(0.3f, e16, fmaf(0.3f, e4, 0.4f * e));
            float kw = K * ywb[r];
            den += kw;
            aK[8 + r] = (_Float16)kw;
        }

        const v16h* fb = fbb + (size_t)mt * 4 * 32;
        acc0 = __builtin_amdgcn_wmma_f32_16x16x32_f16(false, aK, false, fb[0],
                                                      (short)0, acc0, false, false);
        acc1 = __builtin_amdgcn_wmma_f32_16x16x32_f16(false, aK, false, fb[32],
                                                      (short)0, acc1, false, false);
        acc2 = __builtin_amdgcn_wmma_f32_16x16x32_f16(false, aK, false, fb[64],
                                                      (short)0, acc2, false, false);
        acc3 = __builtin_amdgcn_wmma_f32_16x16x32_f16(false, aK, false, fb[96],
                                                      (short)0, acc3, false, false);
    }

    // den: lane holds partial for n = col over half the m's; fold halves
    den += __shfl_xor(den, 16, 32);
    float inv[8];
    #pragma unroll
    for (int r = 0; r < 8; ++r) {
        float dd = __shfl(den, 8 * h + r, 32);   // den for output row n = 8h + r
        inv[r] = 1.0f / (dd + 1e-7f);
    }

    // output tile: lane = column c (within 16-col ct tile), VGPR r = row n = 8h + r
    float* op = out + (size_t)(b * N_ + n0) * C_ + col;
    #pragma unroll
    for (int r = 0; r < 8; ++r) {
        int row = 8 * h + r;
        op[(size_t)row * C_ +  0] = acc0[r] * inv[r];
        op[(size_t)row * C_ + 16] = acc1[r] * inv[r];
        op[(size_t)row * C_ + 32] = acc2[r] * inv[r];
        op[(size_t)row * C_ + 48] = acc3[r] * inv[r];
    }
}

extern "C" void kernel_launch(void* const* d_in, const int* in_sizes, int n_in,
                              void* d_out, int out_size, void* d_ws, size_t ws_size,
                              hipStream_t stream) {
    const float* x    = (const float*)d_in[0];
    const float* y    = (const float*)d_in[1];
    const float* yfea = (const float*)d_in[2];
    const float* yw   = (const float*)d_in[3];
    float* out = (float*)d_out;

    char* ws = (char*)d_ws;
    _Float16* feaB = (_Float16*)(ws);
    float*    yA   = (float*)(ws + YA_OFF);
    float*    yaux = (float*)(ws + YAUX_OFF);

    pack_y_kernel<<<(B_ * M_ + 255) / 256, 256, 0, stream>>>(y, yw, yA, yaux);
    pack_fea_kernel<<<(B_ * M_ * C_ + 255) / 256, 256, 0, stream>>>(yfea, feaB);
    nadwat_main_kernel<<<(B_ * NT_) / 4, 128, 0, stream>>>(x, feaB, yA, yaux, out);
}
